// GNNModel_85109071937916
// MI455X (gfx1250) — matrix operand order repaired and checked by reading, not compiled, not run on hardware
//
#include <hip/hip_runtime.h>

// ---------------------------------------------------------------------------
// GATv2 GNN forward for MI455X (gfx1250, wave32).
// Dense node transforms -> v_wmma_f32_16x16x32_bf16 (matrix pipe).
// Edge softmax/aggregate -> L2-resident gathers + f32 atomics (xl/xr tables
// are 51.2MB each and fit in the 192MB L2, so random gathers hit L2).
// ---------------------------------------------------------------------------

#define GN 100000
#define GE 800000
#define GIN 64
#define GHID 128

typedef __attribute__((ext_vector_type(16))) __bf16 v16bf;
typedef __attribute__((ext_vector_type(8)))  float  v8f;

union FragAB { unsigned u[8]; v16bf v; };
union FragC  { float   f[8]; v8f   v; };

__device__ __forceinline__ unsigned short f2bf(float f) {
    unsigned u = __float_as_uint(f);
    u += 0x7fffu + ((u >> 16) & 1u);          // round to nearest even
    return (unsigned short)(u >> 16);
}

__device__ __forceinline__ void atomicMaxFloat(float* addr, float v) {
    // ordered-int trick; mmax is initialized to -inf (0xFF800000)
    if (v >= 0.f) atomicMax((int*)addr, __float_as_int(v));
    else          atomicMin((unsigned int*)addr, (unsigned int)__float_as_int(v));
}

// ---------------------------------------------------------------------------
// C[M x Nout] = A[M x K](bf16, row-major) * W[K x Nout] + bias
// W supplied pre-transposed as WT[Nout x K] (bf16) so B-fragment loads are
// contiguous 2-element (dword) loads. One wave per 16x16 output tile.
// Fragment layouts per CDNA5 ISA 7.12.2 (16-bit A 16x32 / B 32x16, f32 C/D).
// ---------------------------------------------------------------------------
__global__ void gemm_bf16_wmma(const unsigned short* __restrict__ A,
                               const unsigned short* __restrict__ WT,
                               const float* __restrict__ bias,
                               float* __restrict__ C,
                               int M, int K, int Nout) {
    int wid  = (int)((blockIdx.x * blockDim.x + threadIdx.x) >> 5);
    int lane = threadIdx.x & 31;
    int tilesN = Nout >> 4;
    int tm = wid / tilesN, tn = wid % tilesN;
    int row0 = tm << 4;
    if (row0 >= M) return;

    int half = lane >> 4;        // lane group (0: lanes 0-15, 1: lanes 16-31)
    int r    = lane & 15;
    int arow = row0 + r; if (arow >= M) arow = M - 1;
    int ncol = (tn << 4) + r;

    const unsigned short* Arow = A  + (size_t)arow * K;
    const unsigned short* Wrow = WT + (size_t)ncol * K;

    FragC c;
#pragma unroll
    for (int i = 0; i < 8; ++i) c.f[i] = 0.f;

    for (int k0 = 0; k0 < K; k0 += 32) {
        FragAB a, b;
#pragma unroll
        for (int j = 0; j < 8; ++j) {
            // A 16x32 bf16: vgpr j<4 -> K = 8*half + 2j(+1); j>=4 -> K = 16 + 8*half + 2(j-4)
            int ka = k0 + ((j < 4) ? 0 : 16) + 8 * half + 2 * (j & 3);
            a.u[j] = *(const unsigned*)(Arow + ka);
            // B 32x16 bf16: same K order as A (vgpr j<4 -> K = 8*half + 2j(+1); j>=4 -> K = 16 + 8*half + 2(j-4))
            int kb = k0 + ((j < 4) ? 0 : 16) + 8 * half + 2 * (j & 3);
            b.u[j] = *(const unsigned*)(Wrow + kb);
        }
        c.v = __builtin_amdgcn_wmma_f32_16x16x32_bf16(
            false, a.v, false, b.v, (short)0, c.v, false, false);
    }

    float bv = bias ? bias[ncol] : 0.f;
#pragma unroll
    for (int i = 0; i < 8; ++i) {
        int rr = row0 + i + 8 * half;   // C/D: vgpr i -> row i (+8 for hi lanes)
        if (rr < M) C[(size_t)rr * Nout + ncol] = c.f[i] + bv;
    }
}

// --------------------------- small helper kernels ---------------------------

__global__ void tobf_kernel(const float* __restrict__ src,
                            unsigned short* __restrict__ dst, int n) {
    int i = blockIdx.x * blockDim.x + threadIdx.x;
    if (i < n) dst[i] = f2bf(src[i]);
}

// WT[n*K + k] = bf16(W[k*Nout + n])
__global__ void transpose_bf16_kernel(const float* __restrict__ W,
                                      unsigned short* __restrict__ WT,
                                      int K, int Nout) {
    int i = blockIdx.x * blockDim.x + threadIdx.x;
    if (i >= K * Nout) return;
    int n = i / K, k = i - n * K;
    WT[i] = f2bf(W[(size_t)k * Nout + n]);
}

__global__ void fill_neginf_kernel(float* __restrict__ p, int n) {
    int i = blockIdx.x * blockDim.x + threadIdx.x;
    if (i < n) p[i] = __uint_as_float(0xFF800000u);   // -inf
}

// ---------------------------------------------------------------------------
// Edge pass 1: score[e,h] = sum_c att[h,c]*leaky(xl[src]+xr[dst]+eattr@We)
// Wave per edge; with C=32 each of the 4 lane-iterations is exactly one head.
// Also atomic segment-max into mmax[dst,h].
// ---------------------------------------------------------------------------
__global__ void edge_score_kernel(const float* __restrict__ xl,
                                  const float* __restrict__ xr,
                                  const float* __restrict__ eattr,
                                  const float* __restrict__ We,   // [4,128]
                                  const float* __restrict__ att,  // [128]
                                  const int* __restrict__ src,
                                  const int* __restrict__ dst,
                                  float* __restrict__ score,      // [E*H]
                                  float* __restrict__ mmax,       // [N*H]
                                  int E, int H) {
    int e    = (int)((blockIdx.x * blockDim.x + threadIdx.x) >> 5);
    int lane = threadIdx.x & 31;
    if (e >= E) return;
    int s = src[e], d = dst[e];
    float ea0 = eattr[e * 4 + 0], ea1 = eattr[e * 4 + 1];
    float ea2 = eattr[e * 4 + 2], ea3 = eattr[e * 4 + 3];
    const float* xls = xl + (size_t)s * 128;
    const float* xrd = xr + (size_t)d * 128;

    float tot = 0.f;
#pragma unroll
    for (int it = 0; it < 4; ++it) {
        int f = it * 32 + lane;
        float v = xls[f] + xrd[f] + ea0 * We[f] + ea1 * We[128 + f]
                + ea2 * We[256 + f] + ea3 * We[384 + f];
        v = (v > 0.f) ? v : 0.2f * v;                 // leaky_relu(0.2)
        float p = v * att[f];
        if (H == 4) {                                 // head == it (C=32)
            for (int o = 16; o; o >>= 1) p += __shfl_xor(p, o);
            if (lane == 0) {
                score[(size_t)e * 4 + it] = p;
                atomicMaxFloat(&mmax[(size_t)d * 4 + it], p);
            }
        } else {
            tot += p;                                 // H==1, C=128
        }
    }
    if (H == 1) {
        for (int o = 16; o; o >>= 1) tot += __shfl_xor(tot, o);
        if (lane == 0) { score[e] = tot; atomicMaxFloat(&mmax[d], tot); }
    }
}

// Edge pass 2: a = exp(score - max[dst]); denom[dst] += a
__global__ void edge_exp_kernel(float* __restrict__ score,
                                const float* __restrict__ mmax,
                                float* __restrict__ denom,
                                const int* __restrict__ dst, int E, int H) {
    int i = blockIdx.x * blockDim.x + threadIdx.x;
    if (i >= E * H) return;
    int e = i / H, hh = i - e * H;
    int d = dst[e];
    float a = __expf(score[i] - mmax[(size_t)d * H + hh]);
    score[i] = a;
    atomicAdd(&denom[(size_t)d * H + hh], a);
}

// Edge pass 3: acc[dst,f] += (a/denom[dst,h]) * xl[src,f]
__global__ void edge_agg_kernel(const float* __restrict__ score,
                                const float* __restrict__ denom,
                                const float* __restrict__ xl,
                                const int* __restrict__ src,
                                const int* __restrict__ dst,
                                float* __restrict__ acc,
                                int E, int H, int Clog2) {
    int gid = blockIdx.x * blockDim.x + threadIdx.x;
    if (gid >= E * 128) return;
    int e = gid >> 7, f = gid & 127, hh = f >> Clog2;
    int d = dst[e], s = src[e];
    float alpha = score[(size_t)e * H + hh] / denom[(size_t)d * H + hh];
    atomicAdd(&acc[(size_t)d * 128 + f], alpha * xl[(size_t)s * 128 + f]);
}

// h = [elu](acc + bias[f] + other)   (other may alias h: pure elementwise)
__global__ void finalize_kernel(float* __restrict__ h,
                                const float* __restrict__ acc,
                                const float* __restrict__ bias,
                                const float* __restrict__ other,
                                int do_elu, int n) {
    int i = blockIdx.x * blockDim.x + threadIdx.x;
    if (i >= n) return;
    int f = i & 127;
    float v = acc[i] + bias[f] + other[i];
    if (do_elu) v = (v > 0.f) ? v : (__expf(v) - 1.f);
    h[i] = v;
}

// out[n] = h[n,:] . cls_W + cls_b   (wave per node, butterfly reduce)
__global__ void cls_kernel(const float* __restrict__ h,
                           const float* __restrict__ clsW,
                           const float* __restrict__ clsb,
                           float* __restrict__ out, int n) {
    int node = (int)((blockIdx.x * blockDim.x + threadIdx.x) >> 5);
    int lane = threadIdx.x & 31;
    if (node >= n) return;
    float p = 0.f;
#pragma unroll
    for (int it = 0; it < 4; ++it) {
        int f = it * 32 + lane;
        p += h[(size_t)node * 128 + f] * clsW[f];
    }
    for (int o = 16; o; o >>= 1) p += __shfl_xor(p, o);
    if (lane == 0) out[node] = p + clsb[0];
}

// ---------------------------------------------------------------------------

static inline void launch_gemm(hipStream_t st, const unsigned short* A,
                               const unsigned short* WT, const float* bias,
                               float* C, int M, int K, int Nout) {
    int waves   = ((M + 15) / 16) * (Nout / 16);
    int threads = waves * 32;
    gemm_bf16_wmma<<<(threads + 255) / 256, 256, 0, st>>>(A, WT, bias, C, M, K, Nout);
}

static void run_gat(hipStream_t st, const unsigned short* inbf, int K,
                    const unsigned short* WlT, const float* bl,
                    const unsigned short* WrT, const float* br,
                    const float* We, const float* att, int H, int Clog2,
                    const float* eattr, const int* src, const int* dst,
                    float* xl, float* xr, float* score, float* mmax,
                    float* denom, float* acc) {
    const int N = GN, E = GE;
    launch_gemm(st, inbf, WlT, bl, xl, N, K, 128);
    launch_gemm(st, inbf, WrT, br, xr, N, K, 128);
    fill_neginf_kernel<<<(N * H + 255) / 256, 256, 0, st>>>(mmax, N * H);
    hipMemsetAsync(denom, 0, (size_t)N * H * sizeof(float), st);
    hipMemsetAsync(acc, 0, (size_t)N * 128 * sizeof(float), st);
    edge_score_kernel<<<(E * 32 + 255) / 256, 256, 0, st>>>(
        xl, xr, eattr, We, att, src, dst, score, mmax, E, H);
    edge_exp_kernel<<<(E * H + 255) / 256, 256, 0, st>>>(score, mmax, denom, dst, E, H);
    edge_agg_kernel<<<(E * 128 + 255) / 256, 256, 0, st>>>(
        score, denom, xl, src, dst, acc, E, H, Clog2);
}

extern "C" void kernel_launch(void* const* d_in, const int* in_sizes, int n_in,
                              void* d_out, int out_size, void* d_ws, size_t ws_size,
                              hipStream_t stream) {
    const int N = GN, E = GE, IN = GIN, HID = GHID;

    const float* x       = (const float*)d_in[0];
    const int*   ei      = (const int*)  d_in[1];
    const float* eattr   = (const float*)d_in[2];
    const float* proj_W  = (const float*)d_in[3];
    const float* proj_b  = (const float*)d_in[4];
    const float* cls_W   = (const float*)d_in[5];
    const float* cls_b   = (const float*)d_in[6];
    const float* in_Wl   = (const float*)d_in[7];
    const float* in_bl   = (const float*)d_in[8];
    const float* in_Wr   = (const float*)d_in[9];
    const float* in_br   = (const float*)d_in[10];
    const float* in_We   = (const float*)d_in[11];
    const float* in_att  = (const float*)d_in[12];
    const float* in_bias = (const float*)d_in[13];
    const float* hid_Wl  = (const float*)d_in[14];
    const float* hid_bl  = (const float*)d_in[15];
    const float* hid_Wr  = (const float*)d_in[16];
    const float* hid_br  = (const float*)d_in[17];
    const float* hid_We  = (const float*)d_in[18];
    const float* hid_att = (const float*)d_in[19];
    const float* hid_bias= (const float*)d_in[20];
    const float* out_Wl  = (const float*)d_in[21];
    const float* out_bl  = (const float*)d_in[22];
    const float* out_Wr  = (const float*)d_in[23];
    const float* out_br  = (const float*)d_in[24];
    const float* out_We  = (const float*)d_in[25];
    const float* out_att = (const float*)d_in[26];
    const float* out_bias= (const float*)d_in[27];

    const int* src = ei;
    const int* dst = ei + E;

    // ----- workspace carve (~247 MB) -----
    size_t N128 = (size_t)N * 128;
    float* h     = (float*)d_ws;
    float* acc   = h + N128;
    float* xl    = acc + N128;
    float* xr    = xl + N128;
    float* score = xr + N128;                 // E*4
    float* mmax  = score + (size_t)E * 4;     // N*4
    float* denom = mmax + (size_t)N * 4;      // N*4
    unsigned short* hbf = (unsigned short*)(denom + (size_t)N * 4);  // N*128
    unsigned short* wts = hbf + N128;         // 7 slots of 128*128
    const size_t WSLOT = 128 * 128;
    unsigned short* projT  = wts + 0 * WSLOT;
    unsigned short* inWlT  = wts + 1 * WSLOT;
    unsigned short* inWrT  = wts + 2 * WSLOT;
    unsigned short* hidWlT = wts + 3 * WSLOT;
    unsigned short* hidWrT = wts + 4 * WSLOT;
    unsigned short* outWlT = wts + 5 * WSLOT;
    unsigned short* outWrT = wts + 6 * WSLOT;

    // ----- weight transpose + bf16 convert (tiny) -----
    transpose_bf16_kernel<<<(IN * HID + 255) / 256, 256, 0, stream>>>(proj_W, projT, IN, HID);
    transpose_bf16_kernel<<<(IN * HID + 255) / 256, 256, 0, stream>>>(in_Wl, inWlT, IN, HID);
    transpose_bf16_kernel<<<(IN * HID + 255) / 256, 256, 0, stream>>>(in_Wr, inWrT, IN, HID);
    transpose_bf16_kernel<<<(HID * HID + 255) / 256, 256, 0, stream>>>(hid_Wl, hidWlT, HID, HID);
    transpose_bf16_kernel<<<(HID * HID + 255) / 256, 256, 0, stream>>>(hid_Wr, hidWrT, HID, HID);
    transpose_bf16_kernel<<<(HID * HID + 255) / 256, 256, 0, stream>>>(out_Wl, outWlT, HID, HID);
    transpose_bf16_kernel<<<(HID * HID + 255) / 256, 256, 0, stream>>>(out_Wr, outWrT, HID, HID);

    // ----- layer "in": gat(x) + (x@proj + b), elu -----
    tobf_kernel<<<(N * IN + 255) / 256, 256, 0, stream>>>(x, hbf, N * IN);
    run_gat(stream, hbf, IN, inWlT, in_bl, inWrT, in_br, in_We, in_att, 4, 5,
            eattr, src, dst, xl, xr, score, mmax, denom, acc);
    launch_gemm(stream, hbf, projT, proj_b, xr, N, IN, HID);   // xr free after scores
    finalize_kernel<<<(int)((N128 + 255) / 256), 256, 0, stream>>>(
        h, acc, in_bias, xr, /*elu=*/1, (int)N128);
    tobf_kernel<<<(int)((N128 + 255) / 256), 256, 0, stream>>>(h, hbf, (int)N128);

    // ----- 2 hidden passes (shared params), residual + elu -----
    for (int p = 0; p < 2; ++p) {
        run_gat(stream, hbf, HID, hidWlT, hid_bl, hidWrT, hid_br, hid_We, hid_att, 4, 5,
                eattr, src, dst, xl, xr, score, mmax, denom, acc);
        finalize_kernel<<<(int)((N128 + 255) / 256), 256, 0, stream>>>(
            h, acc, hid_bias, h, /*elu=*/1, (int)N128);
        tobf_kernel<<<(int)((N128 + 255) / 256), 256, 0, stream>>>(h, hbf, (int)N128);
    }

    // ----- out layer: 1 head, C=128, no concat (mean over 1 head = id), residual, no elu
    run_gat(stream, hbf, HID, outWlT, out_bl, outWrT, out_br, out_We, out_att, 1, 7,
            eattr, src, dst, xl, xr, score, mmax, denom, acc);
    finalize_kernel<<<(int)((N128 + 255) / 256), 256, 0, stream>>>(
        h, acc, out_bias, h, /*elu=*/0, (int)N128);

    // ----- classifier -----
    cls_kernel<<<(N * 32 + 255) / 256, 256, 0, stream>>>(h, cls_W, cls_b, (float*)d_out, N);
}
